// LinearLowbit_84696755077628
// MI455X (gfx1250) — compile-verified, matrix operand order checked by reading, not run on hardware
//
#include <hip/hip_runtime.h>
#include <hip/hip_bf16.h>

// ---------------------------------------------------------------------------
// Types for CDNA5 WMMA fragments / TDM descriptors
// ---------------------------------------------------------------------------
typedef __attribute__((ext_vector_type(2)))  float    v2f;
typedef __attribute__((ext_vector_type(8)))  float    v8f;
typedef __attribute__((ext_vector_type(8)))  __bf16   v8bf;
typedef __attribute__((ext_vector_type(16))) __bf16   v16bf;
typedef __attribute__((ext_vector_type(4)))  unsigned v4u;
typedef __attribute__((ext_vector_type(8)))  int      v8i;
typedef __attribute__((ext_vector_type(4)))  int      v4i;

#define EPSQ 1e-8f

// ---------------------------------------------------------------------------
// fp4 e2m1 quant/dequant (per-tensor scale = amax/6, clamped to EPS)
// ---------------------------------------------------------------------------
__device__ __forceinline__ float fp4_qdq_dev(float x, float s) {
  float y = x / s;
  float a = fminf(fabsf(y), 6.0f);
  float lv = 0.0f;
  lv = (a > 0.25f) ? 0.5f : lv;
  lv = (a > 0.75f) ? 1.0f : lv;
  lv = (a > 1.25f) ? 1.5f : lv;
  lv = (a > 1.75f) ? 2.0f : lv;
  lv = (a > 2.5f)  ? 3.0f : lv;
  lv = (a > 3.5f)  ? 4.0f : lv;
  lv = (a > 5.0f)  ? 6.0f : lv;
  return (y < 0.0f ? -lv : lv) * s;
}

__device__ __forceinline__ float scale_from_amax(const unsigned* p) {
  return fmaxf(__uint_as_float(*p) * (1.0f / 6.0f), EPSQ);
}

// ---------------------------------------------------------------------------
// Tensor Data Mover: 2D bf16 tile (tile0 elems x tile1 rows) global -> LDS.
// D# per CDNA5 ISA 8.3/8.4. LDS rows padded: 16 DWORDs data + 4 DWORDs pad
// (pad_interval enc 3 = 16 DW, pad_amount enc 3 = 4 DW) -> 80B LDS row stride,
// which puts the 16 fragment rows in distinct bank groups.
// NOTE: this toolchain's builtin is the 6-arg (clang-23 / therock) form:
//       (v4u g0, v8i g1, v4i g2, v4i g3, v8i g4, i32 cpol)
// ---------------------------------------------------------------------------
__device__ __forceinline__ void tdm_load_tile_bf16(const __bf16* gptr, unsigned lds_off,
                                                   unsigned dim0_rem, unsigned dim1_rem,
                                                   unsigned tile0, unsigned tile1,
                                                   unsigned stride0) {
  unsigned long long ga = (unsigned long long)gptr;
  v4u g0;
  g0[0] = 1u;                                                   // count=1 (user mode)
  g0[1] = lds_off;                                              // lds_addr [63:32]
  g0[2] = (unsigned)ga;                                         // global_addr[31:0]
  g0[3] = (unsigned)((ga >> 32) & 0x01ffffffull) | (2u << 30);  // ga[56:32] | type=2
  v8i g1;
  g1[0] = (int)((1u << 16) | (1u << 20) | (3u << 22) | (3u << 25)); // 2B, pad 4DW/16DW
  g1[1] = (int)((dim0_rem & 0xffffu) << 16);                    // tensor_dim0[15:0]
  g1[2] = (int)(((dim0_rem >> 16) & 0xffffu) | ((dim1_rem & 0xffffu) << 16));
  g1[3] = (int)(((dim1_rem >> 16) & 0xffffu) | (tile0 << 16));  // tile_dim0
  g1[4] = (int)(tile1 & 0xffffu);                               // tile_dim1 (tile_dim2=0)
  g1[5] = (int)stride0;                                         // tensor_dim0_stride[31:0]
  g1[6] = 0;
  g1[7] = 0;
  v4i z4 = {0, 0, 0, 0};
  v8i z8 = {0, 0, 0, 0, 0, 0, 0, 0};
  __builtin_amdgcn_tensor_load_to_lds(g0, g1, z4, z4, z8, 0);
}

// ---------------------------------------------------------------------------
// init: zero the 4 amax scalars (amaxU, amaxV, amaxRes, amaxW)
// ---------------------------------------------------------------------------
__global__ void init_scalars(unsigned* p) {
  if (threadIdx.x < 4) p[threadIdx.x] = 0u;
}

// ---------------------------------------------------------------------------
// Tall-skinny fp32 GEMM via V_WMMA_F32_16X16X4_F32.
// C[Mo,Nd] = A @ B, B[Kd,Nd] row-major (ldb = Nd).
// AT=false: A[row*lda + k]; AT=true: A[k*lda + row].
// ---------------------------------------------------------------------------
template <bool AT>
__global__ void __launch_bounds__(128)
rank_gemm(float* __restrict__ C, const float* __restrict__ A,
          const float* __restrict__ B, int Mo, int Kd, int Nd, int lda) {
  int gwave = (blockIdx.x * 128 + threadIdx.x) >> 5;
  int lane  = threadIdx.x & 31;
  int tilesN = Nd >> 4;
  int tm = gwave / tilesN;
  int tn = gwave - tm * tilesN;
  if (tm >= (Mo >> 4)) return;           // wave-uniform exit
  int lr   = lane & 15;
  int row  = tm * 16 + lr;
  int col  = tn * 16 + lr;
  int koff = (lane >> 4) << 1;           // 0 or 2

  v8f acc = {};
  for (int k = 0; k < Kd; k += 4) {
    v2f a, b;
    if (AT) {
      a[0] = A[(long)(k + koff)     * lda + row];
      a[1] = A[(long)(k + koff + 1) * lda + row];
    } else {
      a[0] = A[(long)row * lda + k + koff];
      a[1] = A[(long)row * lda + k + koff + 1];
    }
    b[0] = B[(long)(k + koff)     * Nd + col];
    b[1] = B[(long)(k + koff + 1) * Nd + col];
    acc = __builtin_amdgcn_wmma_f32_16x16x4_f32(false, a, false, b,
                                                (short)0, acc, false, false);
  }
  int r0 = tm * 16 + ((lane >> 4) << 3);
#pragma unroll
  for (int j = 0; j < 8; ++j)
    C[(long)(r0 + j) * Nd + col] = acc[j];
}

// ---------------------------------------------------------------------------
// Split-K Gram: Gpart[chunk] = P[k0:k0+klen,:]^T @ P[k0:k0+klen,:]  (WMMA f32)
// then deterministic fixed-order reduction over chunks.
// ---------------------------------------------------------------------------
__global__ void __launch_bounds__(128)
gram_part(float* __restrict__ Gp, const float* __restrict__ P,
          int Mlen, int R, int nch) {
  int chunk = blockIdx.y;
  int klen  = Mlen / nch;
  int k0    = chunk * klen;
  int gwave = (blockIdx.x * 128 + threadIdx.x) >> 5;
  int lane  = threadIdx.x & 31;
  int tilesN = R >> 4;
  int tm = gwave / tilesN;
  int tn = gwave - tm * tilesN;
  if (tm >= (R >> 4)) return;
  int lr   = lane & 15;
  int row  = tm * 16 + lr;
  int col  = tn * 16 + lr;
  int koff = (lane >> 4) << 1;

  v8f acc = {};
  for (int k = 0; k < klen; k += 4) {
    v2f a, b;
    a[0] = P[(long)(k0 + k + koff)     * R + row];
    a[1] = P[(long)(k0 + k + koff + 1) * R + row];
    b[0] = P[(long)(k0 + k + koff)     * R + col];
    b[1] = P[(long)(k0 + k + koff + 1) * R + col];
    acc = __builtin_amdgcn_wmma_f32_16x16x4_f32(false, a, false, b,
                                                (short)0, acc, false, false);
  }
  float* out = Gp + (long)chunk * R * R;
  int r0 = tm * 16 + ((lane >> 4) << 3);
#pragma unroll
  for (int j = 0; j < 8; ++j)
    out[(long)(r0 + j) * R + col] = acc[j];
}

__global__ void gram_reduce(float* __restrict__ G, const float* __restrict__ Gp,
                            int n, int nch) {
  int t = blockIdx.x * blockDim.x + threadIdx.x;
  if (t >= n) return;
  float acc = 0.0f;
  for (int c = 0; c < nch; ++c) acc += Gp[(long)c * n + t];
  G[t] = acc;
}

// ---------------------------------------------------------------------------
// Cholesky of Gram (G = L L^T), then Rinv = (L^T)^{-1} = (L^{-1})^T.
// Single block, R (<=64) threads. CholeskyQR: U = P @ Rinv.
// ---------------------------------------------------------------------------
__global__ void __launch_bounds__(64)
chol_inv_kernel(float* __restrict__ Rinv, const float* __restrict__ G, int R) {
  __shared__ float A[64 * 64];
  __shared__ float Bv[64 * 64];
  int t = threadIdx.x;
  for (int j = t; j < R * R; j += 64) A[j] = G[j];
  __syncthreads();
  for (int k = 0; k < R; ++k) {
    if (t == 0) A[k * R + k] = sqrtf(fmaxf(A[k * R + k], 1e-20f));
    __syncthreads();
    if (t > k && t < R) A[t * R + k] /= A[k * R + k];
    __syncthreads();
    if (t > k && t < R) {
      float lik = A[t * R + k];
      for (int j = k + 1; j <= t; ++j) A[t * R + j] -= lik * A[j * R + k];
    }
    __syncthreads();
  }
  if (t < R) {
    for (int i = 0; i < R; ++i) Bv[i * R + t] = 0.0f;
    Bv[t * R + t] = 1.0f / A[t * R + t];
    for (int i = t + 1; i < R; ++i) {
      float s = 0.0f;
      for (int k2 = t; k2 < i; ++k2) s += A[i * R + k2] * Bv[k2 * R + t];
      Bv[i * R + t] = -s / A[i * R + i];
    }
  }
  __syncthreads();
  for (int idx = t; idx < R * R; idx += 64) {
    int r = idx / R, c = idx - r * R;
    Rinv[idx] = (c >= r) ? Bv[c * R + r] : 0.0f;
  }
}

// ---------------------------------------------------------------------------
// Column L2 norms of W [Dn, R] -> S[r]; one block per column
// ---------------------------------------------------------------------------
__global__ void __launch_bounds__(256)
colnorm_kernel(float* __restrict__ S, const float* __restrict__ W, int Dn, int R) {
  int r = blockIdx.x;
  float acc = 0.0f;
  for (int d = threadIdx.x; d < Dn; d += blockDim.x) {
    float v = W[(long)d * R + r];
    acc += v * v;
  }
  __shared__ float red[256];
  red[threadIdx.x] = acc;
  __syncthreads();
  for (int s = 128; s > 0; s >>= 1) {
    if (threadIdx.x < s) red[threadIdx.x] += red[threadIdx.x + s];
    __syncthreads();
  }
  if (threadIdx.x == 0) S[r] = sqrtf(red[0]);
}

__global__ void vnorm_kernel(float* __restrict__ V, const float* __restrict__ W,
                             const float* __restrict__ S, int n, int R) {
  int t = blockIdx.x * blockDim.x + threadIdx.x;
  if (t >= n) return;
  V[t] = W[t] / (S[t % R] + EPSQ);
}

// ---------------------------------------------------------------------------
// |x| max reduction -> atomicMax on float bits (non-negative => uint order ok)
// ---------------------------------------------------------------------------
__global__ void __launch_bounds__(256)
absmax_kernel(unsigned* __restrict__ out, const float* __restrict__ X, long n) {
  float m = 0.0f;
  for (long i = (long)blockIdx.x * blockDim.x + threadIdx.x; i < n;
       i += (long)gridDim.x * blockDim.x)
    m = fmaxf(m, fabsf(X[i]));
  __shared__ float red[256];
  red[threadIdx.x] = m;
  __syncthreads();
  for (int s = 128; s > 0; s >>= 1) {
    if (threadIdx.x < s) red[threadIdx.x] = fmaxf(red[threadIdx.x], red[threadIdx.x + s]);
    __syncthreads();
  }
  if (threadIdx.x == 0) atomicMax(out, __float_as_uint(red[0]));
}

// ---------------------------------------------------------------------------
// US = U * S   and   UqS = fp4_qdq(U) * S
// ---------------------------------------------------------------------------
__global__ void prep_u(float* __restrict__ US, float* __restrict__ UqS,
                       const float* __restrict__ U, const float* __restrict__ S,
                       const unsigned* __restrict__ amaxU, int n, int R) {
  int t = blockIdx.x * blockDim.x + threadIdx.x;
  if (t >= n) return;
  float s  = scale_from_amax(amaxU);
  float sv = S[t % R];
  float u  = U[t];
  US[t]  = u * sv;
  UqS[t] = fp4_qdq_dev(u, s) * sv;
}

__global__ void prep_v(float* __restrict__ Vq, const float* __restrict__ V,
                       const unsigned* __restrict__ amaxV, int n) {
  int t = blockIdx.x * blockDim.x + threadIdx.x;
  if (t >= n) return;
  Vq[t] = fp4_qdq_dev(V[t], scale_from_amax(amaxV));
}

// ---------------------------------------------------------------------------
// amax over residual  res = x - US @ V^T  (WMMA f32 tiles, rank-R)
// ---------------------------------------------------------------------------
__global__ void __launch_bounds__(256)
resid_amax(unsigned* __restrict__ outAmax, const float* __restrict__ X,
           const float* __restrict__ US, const float* __restrict__ V,
           int Dn, int R) {
  int gwave = (blockIdx.x * 256 + threadIdx.x) >> 5;
  int lane  = threadIdx.x & 31;
  int tilesN = Dn >> 4;
  int tm = gwave / tilesN;
  int tn = gwave - tm * tilesN;
  int lr   = lane & 15;
  int row  = tm * 16 + lr;
  int col  = tn * 16 + lr;
  int koff = (lane >> 4) << 1;

  v8f ker = {};
  for (int k = 0; k < R; k += 4) {
    v2f a = *(const v2f*)&US[(long)row * R + k + koff];
    v2f b = *(const v2f*)&V[(long)col * R + k + koff];
    ker = __builtin_amdgcn_wmma_f32_16x16x4_f32(false, a, false, b,
                                                (short)0, ker, false, false);
  }
  int r0 = tm * 16 + ((lane >> 4) << 3);
  float mx = 0.0f;
#pragma unroll
  for (int j = 0; j < 8; ++j)
    mx = fmaxf(mx, fabsf(X[(long)(r0 + j) * Dn + col] - ker[j]));

  __shared__ float red[256];
  red[threadIdx.x] = mx;
  __syncthreads();
  for (int s = 128; s > 0; s >>= 1) {
    if (threadIdx.x < s) red[threadIdx.x] = fmaxf(red[threadIdx.x], red[threadIdx.x + s]);
    __syncthreads();
  }
  if (threadIdx.x == 0) atomicMax(outAmax, __float_as_uint(red[0]));
}

// ---------------------------------------------------------------------------
// xq = UqS @ Vq^T + fp4_qdq(x - US @ V^T)  -> bf16
// ---------------------------------------------------------------------------
__global__ void __launch_bounds__(256)
build_xq(__bf16* __restrict__ XQ, const float* __restrict__ X,
         const float* __restrict__ US, const float* __restrict__ V,
         const float* __restrict__ UqS, const float* __restrict__ Vq,
         const unsigned* __restrict__ amaxRes, int Dn, int R) {
  int gwave = (blockIdx.x * 256 + threadIdx.x) >> 5;
  int lane  = threadIdx.x & 31;
  int tilesN = Dn >> 4;
  int tm = gwave / tilesN;
  int tn = gwave - tm * tilesN;
  int lr   = lane & 15;
  int row  = tm * 16 + lr;
  int col  = tn * 16 + lr;
  int koff = (lane >> 4) << 1;

  v8f ker = {}, rec = {};
  for (int k = 0; k < R; k += 4) {
    v2f a1 = *(const v2f*)&US[(long)row * R + k + koff];
    v2f b1 = *(const v2f*)&V[(long)col * R + k + koff];
    ker = __builtin_amdgcn_wmma_f32_16x16x4_f32(false, a1, false, b1,
                                                (short)0, ker, false, false);
    v2f a2 = *(const v2f*)&UqS[(long)row * R + k + koff];
    v2f b2 = *(const v2f*)&Vq[(long)col * R + k + koff];
    rec = __builtin_amdgcn_wmma_f32_16x16x4_f32(false, a2, false, b2,
                                                (short)0, rec, false, false);
  }
  float s = scale_from_amax(amaxRes);
  int r0 = tm * 16 + ((lane >> 4) << 3);
#pragma unroll
  for (int j = 0; j < 8; ++j) {
    long idx = (long)(r0 + j) * Dn + col;
    float v = rec[j] + fp4_qdq_dev(X[idx] - ker[j], s);
    XQ[idx] = (__bf16)v;
  }
}

// ---------------------------------------------------------------------------
// wq = fp4_qdq(weight) -> bf16
// ---------------------------------------------------------------------------
__global__ void __launch_bounds__(256)
quant_w(__bf16* __restrict__ WQ, const float* __restrict__ W,
        const unsigned* __restrict__ amaxW, long n) {
  float s = scale_from_amax(amaxW);
  for (long i = (long)blockIdx.x * blockDim.x + threadIdx.x; i < n;
       i += (long)gridDim.x * blockDim.x)
    WQ[i] = (__bf16)fp4_qdq_dev(W[i], s);
}

// ---------------------------------------------------------------------------
// Big GEMM: out[M,Dout] = XQ @ WQ^T + bias.  bf16 WMMA 16x16x32.
// Block = 256 threads = 8 waves (4x2), block tile 128x128, K-chunk 32.
// TDM double-buffered staging: wave0 waits TENSORcnt for chunk k, then issues
// tensor_load_to_lds for the A/B tiles of chunk k+1, then all waves consume the
// padded LDS tiles (80B row stride -> conflict-free ds_load_b128) with WMMA.
// ---------------------------------------------------------------------------
#define LDS_ROW 40   // bf16 elems per LDS row (64B data + 16B TDM pad)

__global__ void __launch_bounds__(256)
big_gemm(float* __restrict__ out, const __bf16* __restrict__ Aq,
         const __bf16* __restrict__ Wq, const float* __restrict__ bias,
         int Mlen, int Dn, int Dout) {
  __shared__ __align__(16) __bf16 ldsA[2][128 * LDS_ROW];
  __shared__ __align__(16) __bf16 ldsB[2][128 * LDS_ROW];

  const int lane = threadIdx.x & 31;
  const int wave = threadIdx.x >> 5;
  const int wm = wave & 3, wn = wave >> 2;
  const int mBlk = blockIdx.x * 128;
  const int nBlk = blockIdx.y * 128;
  const int lr = lane & 15;
  const int kb = (lane >> 4) * 8;
  const bool issuer = (wave == 0);

  unsigned ldsAoff[2], ldsBoff[2];
  ldsAoff[0] = (unsigned)(unsigned long long)&ldsA[0][0];
  ldsAoff[1] = (unsigned)(unsigned long long)&ldsA[1][0];
  ldsBoff[0] = (unsigned)(unsigned long long)&ldsB[0][0];
  ldsBoff[1] = (unsigned)(unsigned long long)&ldsB[1][0];

  const int NK = Dn / 32;
  if (issuer) {
    tdm_load_tile_bf16(Aq + (long)mBlk * Dn, ldsAoff[0],
                       (unsigned)Dn, (unsigned)(Mlen - mBlk), 32u, 128u, (unsigned)Dn);
    tdm_load_tile_bf16(Wq + (long)nBlk * Dn, ldsBoff[0],
                       (unsigned)Dn, (unsigned)(Dout - nBlk), 32u, 128u, (unsigned)Dn);
  }

  v8f acc[2][4] = {};
  for (int kc = 0; kc < NK; ++kc) {
    const int cur = kc & 1;
    if (issuer) __builtin_amdgcn_s_wait_tensorcnt(0);
    __syncthreads();                         // tiles for `cur` visible to all waves
    if (issuer && (kc + 1 < NK)) {           // prefetch next chunk into other buffer
      int kn = (kc + 1) * 32;
      tdm_load_tile_bf16(Aq + (long)mBlk * Dn + kn, ldsAoff[1 - cur],
                         (unsigned)(Dn - kn), (unsigned)(Mlen - mBlk), 32u, 128u,
                         (unsigned)Dn);
      tdm_load_tile_bf16(Wq + (long)nBlk * Dn + kn, ldsBoff[1 - cur],
                         (unsigned)(Dn - kn), (unsigned)(Dout - nBlk), 32u, 128u,
                         (unsigned)Dn);
    }

    v16bf a[2], b[4];
#pragma unroll
    for (int i = 0; i < 2; ++i) {
      int r = wm * 32 + i * 16 + lr;
      const __bf16* p = &ldsA[cur][r * LDS_ROW + kb];
      v8bf lo = *(const v8bf*)(p);
      v8bf hi = *(const v8bf*)(p + 16);
#pragma unroll
      for (int e = 0; e < 8; ++e) { a[i][e] = lo[e]; a[i][8 + e] = hi[e]; }
    }
#pragma unroll
    for (int j = 0; j < 4; ++j) {
      int r = wn * 64 + j * 16 + lr;
      const __bf16* p = &ldsB[cur][r * LDS_ROW + kb];
      v8bf lo = *(const v8bf*)(p);
      v8bf hi = *(const v8bf*)(p + 16);
#pragma unroll
      for (int e = 0; e < 8; ++e) { b[j][e] = lo[e]; b[j][8 + e] = hi[e]; }
    }
#pragma unroll
    for (int i = 0; i < 2; ++i)
#pragma unroll
      for (int j = 0; j < 4; ++j)
        acc[i][j] = __builtin_amdgcn_wmma_f32_16x16x32_bf16(
            false, a[i], false, b[j], (short)0, acc[i][j], false, false);
    __syncthreads();                         // done reading `cur` before it is reloaded
  }

#pragma unroll
  for (int i = 0; i < 2; ++i) {
    int r0 = mBlk + wm * 32 + i * 16 + ((lane >> 4) << 3);
#pragma unroll
    for (int j = 0; j < 4; ++j) {
      int c = nBlk + wn * 64 + j * 16 + lr;
      float bs = bias[c];
#pragma unroll
      for (int e = 0; e < 8; ++e)
        out[(long)(r0 + e) * Dout + c] = acc[i][j][e] + bs;
    }
  }
}

// ---------------------------------------------------------------------------
// Host-side orchestration
// ---------------------------------------------------------------------------
extern "C" void kernel_launch(void* const* d_in, const int* in_sizes, int n_in,
                              void* d_out, int out_size, void* d_ws, size_t ws_size,
                              hipStream_t stream) {
  const float* x      = (const float*)d_in[0];
  const float* weight = (const float*)d_in[1];
  const float* bias   = (const float*)d_in[2];
  const float* V0     = (const float*)d_in[3];

  const int Dout = in_sizes[2];
  const int Din  = in_sizes[1] / Dout;
  const int M    = in_sizes[0] / Din;
  const int R    = in_sizes[3] / Din;   // 64
  float* out = (float*)d_out;

  char* ws = (char*)d_ws;
  auto carve = [&](size_t bytes) -> char* {
    char* p = ws;
    ws += (bytes + 255) & ~(size_t)255;
    return p;
  };
  const int nch = 64;                                   // Gram split-K chunks
  float*    P     = (float*)carve((size_t)M * R * 4);   // P, later US = U*S
  float*    U     = (float*)carve((size_t)M * R * 4);
  float*    UqS   = (float*)carve((size_t)M * R * 4);
  float*    Wm    = (float*)carve((size_t)Din * R * 4);
  float*    Vc    = (float*)carve((size_t)Din * R * 4);
  float*    Vq    = (float*)carve((size_t)Din * R * 4);
  float*    Gpart = (float*)carve((size_t)nch * R * R * 4);
  float*    Gram  = (float*)carve((size_t)R * R * 4);
  float*    Rinv  = (float*)carve((size_t)R * R * 4);
  float*    Svec  = (float*)carve((size_t)R * 4);
  unsigned* amax  = (unsigned*)carve(4 * sizeof(unsigned));
  __bf16*   XQ    = (__bf16*)carve((size_t)M * Din * 2);
  __bf16*   WQ    = (__bf16*)carve((size_t)Dout * Din * 2);

  init_scalars<<<1, 32, 0, stream>>>(amax);

  // ---- power iteration (NITER = 2), QR via Cholesky-QR -------------------
  const int gramTiles = (R / 16) * (R / 16);            // 16
  for (int it = 0; it < 2; ++it) {
    const float* Vin = (it == 0) ? V0 : Vc;
    int wavesP = (M / 16) * (R / 16);
    rank_gemm<false><<<wavesP / 4, 128, 0, stream>>>(P, x, Vin, M, Din, R, Din);
    gram_part<<<dim3((gramTiles + 3) / 4, nch), 128, 0, stream>>>(Gpart, P, M, R, nch);
    gram_reduce<<<(R * R + 255) / 256, 256, 0, stream>>>(Gram, Gpart, R * R, nch);
    chol_inv_kernel<<<1, 64, 0, stream>>>(Rinv, Gram, R);
    rank_gemm<false><<<wavesP / 4, 128, 0, stream>>>(U, P, Rinv, M, R, R, R);
    int wavesW = (Din / 16) * (R / 16);
    rank_gemm<true><<<wavesW / 4, 128, 0, stream>>>(Wm, x, U, Din, M, R, Din);
    colnorm_kernel<<<R, 256, 0, stream>>>(Svec, Wm, Din, R);
    vnorm_kernel<<<(Din * R) / 256, 256, 0, stream>>>(Vc, Wm, Svec, Din * R, R);
  }

  // ---- fp4 quantization scales -------------------------------------------
  absmax_kernel<<<512, 256, 0, stream>>>(amax + 0, U, (long)M * R);
  absmax_kernel<<<512, 256, 0, stream>>>(amax + 1, Vc, (long)Din * R);
  absmax_kernel<<<1024, 256, 0, stream>>>(amax + 3, weight, (long)Dout * Din);

  prep_u<<<(M * R) / 256, 256, 0, stream>>>(P /*US*/, UqS, U, Svec, amax + 0, M * R, R);
  prep_v<<<(Din * R) / 256, 256, 0, stream>>>(Vq, Vc, amax + 1, Din * R);

  int tiles = (M / 16) * (Din / 16);
  resid_amax<<<tiles / 8, 256, 0, stream>>>(amax + 2, x, P, Vc, Din, R);
  build_xq<<<tiles / 8, 256, 0, stream>>>(XQ, x, P, Vc, UqS, Vq, amax + 2, Din, R);
  quant_w<<<1024, 256, 0, stream>>>(WQ, weight, amax + 3, (long)Dout * Din);

  // ---- main GEMM: out = XQ @ WQ^T + bias (TDM + WMMA bf16) ---------------
  dim3 grid(M / 128, Dout / 128);
  big_gemm<<<grid, 256, 0, stream>>>(out, XQ, WQ, bias, M, Din, Dout);
}
